// EagleAttention_22419729285272
// MI455X (gfx1250) — matrix operand order repaired and checked by reading, not compiled, not run on hardware
//
#include <hip/hip_runtime.h>
#include <math.h>

#define HIDDEN_ 4096
#define NH_ 32
#define NKV_ 8
#define HD_ 128
#define BB_ 2
#define SS_ 2048
#define MTOT_ (BB_*SS_)           // 4096 rows
#define NKVD_ 1024
#define NTOT_ (HIDDEN_ + 2*NKVD_) // 6144 fused QKV cols

#define MBLK 128
#define NBLK 64
#define LDSROW 40                 // 32 data bf16 + 8 pad (80B rows, 16B-aligned)

typedef __attribute__((ext_vector_type(16))) __bf16 v16bf;
typedef __attribute__((ext_vector_type(8)))  __bf16 v8bf;
typedef __attribute__((ext_vector_type(4)))  __bf16 v4bf;
typedef __attribute__((ext_vector_type(8)))  float  v8f;

__device__ __forceinline__ __bf16 to_bf16(float x) { return (__bf16)x; }

__device__ __forceinline__ v8f wmma_bf16(v16bf a, v16bf b, v8f c) {
  return __builtin_amdgcn_wmma_f32_16x16x32_bf16(false, a, false, b, (short)0, c,
                                                 false, false);
}

// ---------- gfx1250 async copy: GLOBAL_LOAD_ASYNC_TO_LDS_B128 (ASYNCcnt) ----------
typedef __attribute__((address_space(3))) __bf16 lds_bf16_t;
__device__ __forceinline__ void async_cp16(__bf16* lds_dst, const __bf16* gsrc) {
  unsigned loff = (unsigned)(unsigned long long)(lds_bf16_t*)lds_dst;
  unsigned long long gaddr = (unsigned long long)gsrc;
  asm volatile("global_load_async_to_lds_b128 %0, %1, off"
               :: "v"(loff), "v"(gaddr) : "memory");
}
__device__ __forceinline__ void wait_async0() {
  asm volatile("s_wait_asynccnt 0x0" ::: "memory");
}

// ---------- bf16 fragment loaders (ISA 7.12.2 layouts), all 16B-aligned loads ----------
__device__ __forceinline__ v16bf frag_from_2x8(const __bf16* p, const __bf16* q) {
  v8bf lo = *(const v8bf*)p;
  v8bf hi = *(const v8bf*)q;
  v16bf r;
#pragma unroll
  for (int i = 0; i < 8; ++i) { r[i] = lo[i]; r[8 + i] = hi[i]; }
  return r;
}
// A 16x32: lane row = L&15, K = {kh..kh+7} U {kh+16..kh+23}, kh = (L>>4)*8
__device__ __forceinline__ v16bf load_frag_a_bf(const __bf16* base, long long ld, int lane) {
  const __bf16* p = base + (size_t)(lane & 15) * ld + ((lane >> 4) << 3);
  return frag_from_2x8(p, p + 16);
}
// B 32x16 (NT): lane col = L&15, K = {kh..kh+7} U {kh+16..kh+23}, kh = (L>>4)*8
__device__ __forceinline__ v16bf load_frag_b_bf(const __bf16* base, long long ld, int lane) {
  const __bf16* p = base + (size_t)(lane & 15) * ld + ((lane >> 4) << 3);
  return frag_from_2x8(p, p + 16);
}

// =====================================================================================
// Kernel 0: fp32 -> bf16 conversion (one pass; removes all v_cvt from GEMM hot loops)
// =====================================================================================
__global__ __launch_bounds__(256) void cvt_f32_to_bf16(
    const float* __restrict__ s, __bf16* __restrict__ d, long long n) {
  long long i = ((long long)blockIdx.x * 256 + threadIdx.x) * 4;
  if (i < n) {
    float4 v = *(const float4*)(s + i);
    v4bf o;
    o[0] = to_bf16(v.x); o[1] = to_bf16(v.y); o[2] = to_bf16(v.z); o[3] = to_bf16(v.w);
    *(v4bf*)(d + i) = o;
  }
}

// =====================================================================================
// LDS-staged, async double-buffered NT GEMM: C[m,n] = sum_k X[m,k]*W[n,k]  (bf16 in,
// f32 accum). Block 256 thr = 8 waves, 128x64 tile, wave = 32x32 (4 wmma / k-step).
// mode 0: fused QKV -> scatter stores Q[B,NH,S,D], K[B,NKV,S,D], V^T[B,NKV,D,S] (bf16)
// mode 1: O-proj    -> fp32 row-major [MTOT, HIDDEN] to d_out
// =====================================================================================
__global__ __launch_bounds__(256) void eagle_gemm_nt(
    const __bf16* __restrict__ X,
    const __bf16* __restrict__ w0, const __bf16* __restrict__ w1,
    const __bf16* __restrict__ w2,
    __bf16* __restrict__ Qb, __bf16* __restrict__ Kb, __bf16* __restrict__ VbT,
    float* __restrict__ fout, int mode) {
  __shared__ __align__(16) __bf16 As[2][MBLK * LDSROW];  // 2 x 10240 B
  __shared__ __align__(16) __bf16 Bs[2][NBLK * LDSROW];  // 2 x  5120 B
  const int tid = threadIdx.x, w = tid >> 5, lane = tid & 31;
  const int m0 = blockIdx.x * MBLK;
  const int gn = blockIdx.y * NBLK;

  const __bf16* W; int nb, which;
  if (mode == 1)                 { W = w0; nb = gn;                   which = 3; }
  else if (gn < HIDDEN_)         { W = w0; nb = gn;                   which = 0; }
  else if (gn < HIDDEN_ + NKVD_) { W = w1; nb = gn - HIDDEN_;         which = 1; }
  else                           { W = w2; nb = gn - HIDDEN_ - NKVD_; which = 2; }

  // one stage = 12KB = 768 x 16B chunks; 3 async b128 per thread
  auto issue = [&](int buf, int k) {
#pragma unroll
    for (int i = 0; i < 2; ++i) {
      const int c = tid + 256 * i, r = c >> 2, cc = c & 3;
      async_cp16(&As[buf][r * LDSROW + cc * 8],
                 X + (size_t)(m0 + r) * HIDDEN_ + k + cc * 8);
    }
    {
      const int r = tid >> 2, cc = tid & 3;
      async_cp16(&Bs[buf][r * LDSROW + cc * 8],
                 W + (size_t)(nb + r) * HIDDEN_ + k + cc * 8);
    }
  };

  v8f acc00 = {}, acc01 = {}, acc10 = {}, acc11 = {};
  const int wm = (w & 3) * 32, wn = (w >> 2) * 32;

  issue(0, 0);
  for (int k = 0; k < HIDDEN_; k += 32) {
    const int buf = (k >> 5) & 1;
    wait_async0();                       // this wave's stage for `buf` landed
    __syncthreads();                     // everyone's stage landed; prev reads done
    if (k + 32 < HIDDEN_) issue(buf ^ 1, k + 32);   // overlap next copy w/ compute
    const __bf16* A  = As[buf];
    const __bf16* Bm = Bs[buf];
    v16bf a0 = load_frag_a_bf(A  + (size_t)wm        * LDSROW, LDSROW, lane);
    v16bf a1 = load_frag_a_bf(A  + (size_t)(wm + 16) * LDSROW, LDSROW, lane);
    v16bf b0 = load_frag_b_bf(Bm + (size_t)wn        * LDSROW, LDSROW, lane);
    v16bf b1 = load_frag_b_bf(Bm + (size_t)(wn + 16) * LDSROW, LDSROW, lane);
    acc00 = wmma_bf16(a0, b0, acc00);
    acc01 = wmma_bf16(a0, b1, acc01);
    acc10 = wmma_bf16(a1, b0, acc10);
    acc11 = wmma_bf16(a1, b1, acc11);
  }

  const int half = lane >> 4, colL = lane & 15;
#pragma unroll
  for (int mi = 0; mi < 2; ++mi) {
    const int row0 = m0 + wm + mi * 16 + half * 8;
#pragma unroll
    for (int ni = 0; ni < 2; ++ni) {
      v8f acc = mi ? (ni ? acc11 : acc10) : (ni ? acc01 : acc00);
      const int c = nb + wn + ni * 16 + colL;
      if (which == 3) {
        float* dst = fout + (size_t)row0 * HIDDEN_ + c;
#pragma unroll
        for (int j = 0; j < 8; ++j) dst[(size_t)j * HIDDEN_] = acc[j];
      } else {
        const int bidx = row0 / SS_, s0 = row0 % SS_;
        const int hh = c >> 7, dd = c & 127;
        if (which == 0) {
          __bf16* dst = Qb + (((size_t)bidx * NH_ + hh) * SS_ + s0) * HD_ + dd;
#pragma unroll
          for (int j = 0; j < 8; ++j) dst[(size_t)j * HD_] = to_bf16(acc[j]);
        } else if (which == 1) {
          __bf16* dst = Kb + (((size_t)bidx * NKV_ + hh) * SS_ + s0) * HD_ + dd;
#pragma unroll
          for (int j = 0; j < 8; ++j) dst[(size_t)j * HD_] = to_bf16(acc[j]);
        } else {
          // V stored TRANSPOSED [b, kvh, d, s]: consecutive j == consecutive s
          __bf16* dst = VbT + (((size_t)bidx * NKV_ + hh) * HD_ + dd) * SS_ + s0;
          v8bf pk;
#pragma unroll
          for (int j = 0; j < 8; ++j) pk[j] = to_bf16(acc[j]);
          *(v8bf*)dst = pk;
        }
      }
    }
  }
}

// =====================================================================================
// RoPE on bf16 Q and K in place. grid.x = B*(NH+NKV)*S, block = 64 (freq pair i).
// =====================================================================================
__global__ __launch_bounds__(64) void eagle_rope(
    __bf16* __restrict__ Qb, __bf16* __restrict__ Kb,
    const long long* __restrict__ pos_ids) {
  const int idx  = blockIdx.x;
  const int s    = idx % SS_;
  const int head = (idx / SS_) % (NH_ + NKV_);
  const int b    = idx / (SS_ * (NH_ + NKV_));
  __bf16* base = (head < NH_)
      ? Qb + (((size_t)b * NH_  + head)         * SS_ + s) * HD_
      : Kb + (((size_t)b * NKV_ + (head - NH_)) * SS_ + s) * HD_;
  const int i = threadIdx.x;                        // 0..63
  const float pos  = (float)pos_ids[s];
  const float invf = __expf(-(float)i * (9.2103403719761836f / 64.0f)); // 10000^(-2i/128)
  float sn, cs;
  __sincosf(pos * invf, &sn, &cs);
  const float x1 = (float)base[i], x2 = (float)base[i + 64];
  base[i]      = to_bf16(x1 * cs - x2 * sn);
  base[i + 64] = to_bf16(x2 * cs + x1 * sn);
}

// =====================================================================================
// Causal flash attention, GQA 4:1, bf16 operands / f32 accum+softmax.
// grid = (S/64, NH, B), block 128 (4 waves); wave = 16 q rows; V read from V^T.
// =====================================================================================
__global__ __launch_bounds__(128) void eagle_flash_attn(
    const __bf16* __restrict__ Q, const __bf16* __restrict__ K,
    const __bf16* __restrict__ VT, __bf16* __restrict__ O) {
  __shared__ __align__(16) __bf16 plds[4][16 * 32];   // 1KB per wave
  const int w = threadIdx.x >> 5, lane = threadIdx.x & 31;
  const int h = blockIdx.y, b = blockIdx.z;
  const int q0  = blockIdx.x * 64 + w * 16;
  const int kvh = h >> 2;

  const __bf16* Qbase  = Q  + (((size_t)b * NH_  + h)   * SS_ + q0) * HD_;
  const __bf16* Kbase  = K  + (((size_t)b * NKV_ + kvh) * SS_) * HD_;
  const __bf16* VTbase = VT + (((size_t)b * NKV_ + kvh) * HD_) * SS_;

  v16bf qf[4];
#pragma unroll
  for (int t = 0; t < 4; ++t) qf[t] = load_frag_a_bf(Qbase + t * 32, HD_, lane);

  v8f acc[8];
#pragma unroll
  for (int nt = 0; nt < 8; ++nt) acc[nt] = (v8f){};
  float m_[8], l_[8];
#pragma unroll
  for (int j = 0; j < 8; ++j) { m_[j] = -1e30f; l_[j] = 0.0f; }

  const int half = lane >> 4, colL = lane & 15;
  const float sc = 0.08838834764831845f;              // 1/sqrt(128)

  const int kend = q0 + 16;
  for (int kt = 0; kt < kend; kt += 32) {
    __builtin_prefetch(Kbase + (size_t)(kt + 32) * HD_, 0, 1);
    // ---- S = Q @ K^T, two 16-key tiles
    v8f s1 = (v8f){}, s2 = (v8f){};
#pragma unroll
    for (int t = 0; t < 4; ++t) {
      v16bf kb0 = load_frag_b_bf(Kbase + (size_t)kt        * HD_ + t * 32, HD_, lane);
      v16bf kb1 = load_frag_b_bf(Kbase + (size_t)(kt + 16) * HD_ + t * 32, HD_, lane);
      s1 = wmma_bf16(qf[t], kb0, s1);
      s2 = wmma_bf16(qf[t], kb1, s2);
    }

    // ---- online softmax
    const int key1 = kt + colL, key2 = kt + 16 + colL;
    float alpha[8];
#pragma unroll
    for (int j = 0; j < 8; ++j) {
      const int qr = q0 + half * 8 + j;
      float v1 = s1[j] * sc; if (key1 > qr) v1 = -1e30f;
      float v2 = s2[j] * sc; if (key2 > qr) v2 = -1e30f;
      float rm = fmaxf(v1, v2);
#pragma unroll
      for (int off = 1; off < 16; off <<= 1) rm = fmaxf(rm, __shfl_xor(rm, off, 32));
      const float mnew = fmaxf(m_[j], rm);
      alpha[j] = __expf(m_[j] - mnew);
      const float p1 = __expf(v1 - mnew);
      const float p2 = __expf(v2 - mnew);
      float rs = p1 + p2;
#pragma unroll
      for (int off = 1; off < 16; off <<= 1) rs += __shfl_xor(rs, off, 32);
      l_[j] = l_[j] * alpha[j] + rs;
      m_[j] = mnew;
      s1[j] = p1; s2[j] = p2;
    }
#pragma unroll
    for (int nt = 0; nt < 8; ++nt)
#pragma unroll
      for (int j = 0; j < 8; ++j) acc[nt][j] *= alpha[j];

    // ---- P (C-layout f32) -> LDS bf16 -> A-fragment
    __bf16* pl = plds[w];
#pragma unroll
    for (int j = 0; j < 8; ++j) {
      const int r = half * 8 + j;
      pl[r * 32 + colL]      = to_bf16(s1[j]);
      pl[r * 32 + 16 + colL] = to_bf16(s2[j]);
    }
    asm volatile("s_wait_dscnt 0" ::: "memory");
    const __bf16* pp = &plds[w][(size_t)colL * 32 + half * 8];
    v16bf pf = frag_from_2x8(pp, pp + 16);

    // ---- O += P @ V  (V^T: contiguous 16B B-frag loads, ld = S)
#pragma unroll
    for (int nt = 0; nt < 8; ++nt) {
      v16bf vf = load_frag_b_bf(VTbase + (size_t)(nt * 16) * SS_ + kt, SS_, lane);
      acc[nt] = wmma_bf16(pf, vf, acc[nt]);
    }
  }

  // ---- normalize + store O[b, s, h*128+d] (bf16 row-major [B*S, 4096])
  __bf16* Obase = O + ((size_t)(b * SS_ + q0)) * HIDDEN_ + h * HD_;
#pragma unroll
  for (int nt = 0; nt < 8; ++nt) {
    const int d = nt * 16 + colL;
#pragma unroll
    for (int j = 0; j < 8; ++j) {
      const int r = half * 8 + j;
      Obase[(size_t)r * HIDDEN_ + d] = to_bf16(acc[nt][j] / l_[j]);
    }
  }
}

// =====================================================================================
extern "C" void kernel_launch(void* const* d_in, const int* in_sizes, int n_in,
                              void* d_out, int out_size, void* d_ws, size_t ws_size,
                              hipStream_t stream) {
  (void)in_sizes; (void)n_in; (void)out_size; (void)ws_size;
  const float*     hidden = (const float*)d_in[0];
  const long long* pos    = (const long long*)d_in[1];
  const float*     wq     = (const float*)d_in[2];
  const float*     wk     = (const float*)d_in[3];
  const float*     wv     = (const float*)d_in[4];
  const float*     wo     = (const float*)d_in[5];
  float*           out    = (float*)d_out;

  // bf16 workspace (~202 MB)
  __bf16* p   = (__bf16*)d_ws;
  const size_t NXW = (size_t)MTOT_ * HIDDEN_;   // 16.78M
  const size_t NKW = (size_t)NKVD_ * HIDDEN_;   //  4.19M
  const size_t NQ  = (size_t)BB_ * NH_  * SS_ * HD_;
  const size_t NKV = (size_t)BB_ * NKV_ * SS_ * HD_;
  __bf16* Xbf = p; p += NXW;
  __bf16* Wqb = p; p += NXW;
  __bf16* Wkb = p; p += NKW;
  __bf16* Wvb = p; p += NKW;
  __bf16* Wob = p; p += NXW;
  __bf16* Qb  = p; p += NQ;
  __bf16* Kb  = p; p += NKV;
  __bf16* VbT = p; p += NKV;
  __bf16* Ob  = p; p += NXW;

  auto cvtGrid = [](size_t n) { return dim3((unsigned)((n / 4 + 255) / 256)); };
  cvt_f32_to_bf16<<<cvtGrid(NXW), 256, 0, stream>>>(hidden, Xbf, (long long)NXW);
  cvt_f32_to_bf16<<<cvtGrid(NXW), 256, 0, stream>>>(wq, Wqb, (long long)NXW);
  cvt_f32_to_bf16<<<cvtGrid(NKW), 256, 0, stream>>>(wk, Wkb, (long long)NKW);
  cvt_f32_to_bf16<<<cvtGrid(NKW), 256, 0, stream>>>(wv, Wvb, (long long)NKW);
  cvt_f32_to_bf16<<<cvtGrid(NXW), 256, 0, stream>>>(wo, Wob, (long long)NXW);

  eagle_gemm_nt<<<dim3(MTOT_ / MBLK, NTOT_ / NBLK), dim3(256), 0, stream>>>(
      Xbf, Wqb, Wkb, Wvb, Qb, Kb, VbT, nullptr, 0);
  eagle_rope<<<dim3(BB_ * (NH_ + NKV_) * SS_), dim3(64), 0, stream>>>(Qb, Kb, pos);
  eagle_flash_attn<<<dim3(SS_ / 64, NH_, BB_), dim3(128), 0, stream>>>(Qb, Kb, VbT, Ob);
  eagle_gemm_nt<<<dim3(MTOT_ / MBLK, HIDDEN_ / NBLK), dim3(256), 0, stream>>>(
      Ob, Wob, nullptr, nullptr, nullptr, nullptr, nullptr, out, 1);
}
